// MultiHeadAttentionVec_79706003079242
// MI455X (gfx1250) — compile-verified, matrix operand order checked by reading, not run on hardware
//
#include <hip/hip_runtime.h>
#include <hip/hip_bf16.h>

typedef __attribute__((ext_vector_type(16))) _Float16 v16h;
typedef __attribute__((ext_vector_type(8)))  float    v8f;

#define S_LEN   2048
#define DMODEL  1024
#define NHEADS  16
#define DK      64
#define BATCH   2
#define NEGVAL  -1e9f

// ---- WMMA fragment index helpers (ISA 7.12.2, 16-bit A 16x32 layout) ----
// element e (0..15) of the v16h frag, K-half kh (= lane>=16):
//   vgpr = e/2 ; K = base(vgpr) + (e&1) + kh*8 ; base = vg<4 ? 2*vg : 16+2*(vg-4)
__device__ __forceinline__ int frag_k(int e, int kh) {
    int vg = e >> 1;
    int base = (vg < 4) ? (vg * 2) : (16 + (vg - 4) * 2);
    return base + (e & 1) + (kh << 3);
}

// A-frag / B-frag gather when the K dimension is contiguous in memory:
// element value = rowptr[K]  (rowptr already includes the per-lane row).
// Per lane this is two contiguous 8-half runs -> compiler emits 2x ds_load_b128.
__device__ __forceinline__ v16h gather_k(const _Float16* rowptr, int kh) {
    v16h r;
#pragma unroll
    for (int e = 0; e < 16; ++e) r[e] = rowptr[frag_k(e, kh)];
    return r;
}

// --------------------------------------------------------------------------
__global__ void cvt_f32_to_f16(const float* __restrict__ src,
                               _Float16* __restrict__ dst, int n) {
    int i = blockIdx.x * blockDim.x + threadIdx.x;
    if (i < n) dst[i] = (_Float16)src[i];
}

// --------------------------------------------------------------------------
// RoPE in-place on f16 q and k, layout [B, S, H*DK]; one thread per (b,s,h,pair)
__global__ void rope_inplace(_Float16* __restrict__ q, _Float16* __restrict__ k,
                             int total) {
    int idx = blockIdx.x * blockDim.x + threadIdx.x;
    if (idx >= total) return;
    int i = idx & 31;               // pair index 0..31
    int h = (idx >> 5) & (NHEADS - 1);
    int s = (idx >> 9) & (S_LEN - 1);
    int b = idx >> 20;
    float inv_freq = __powf(10000.0f, -(float)i / 32.0f);
    float ang = (float)s * inv_freq;
    float sn, cs;
    __sincosf(ang, &sn, &cs);
    size_t base = ((size_t)(b * S_LEN + s)) * DMODEL + h * DK + 2 * i;
    float t1 = (float)q[base], t2 = (float)q[base + 1];
    q[base]     = (_Float16)(t1 * cs - t2 * sn);
    q[base + 1] = (_Float16)(t1 * sn + t2 * cs);
    t1 = (float)k[base]; t2 = (float)k[base + 1];
    k[base]     = (_Float16)(t1 * cs - t2 * sn);
    k[base + 1] = (_Float16)(t1 * sn + t2 * cs);
}

// --------------------------------------------------------------------------
// C[M,N] = A[M,K] * B[N,K]^T   (both f16 row-major), LDS-tiled, WMMA.
// Block: 256 threads = 8 waves computing a 64x128 C tile (wave = 16x64,
// A-frag reused across 4 WMMA).
__global__ void wmma_gemm_nt(const _Float16* __restrict__ A,
                             const _Float16* __restrict__ Bw,
                             void* __restrict__ Cout,
                             int M, int N, int K, int storeHalf) {
    __shared__ _Float16 As[64 * 40];    // 64 rows x 32 k (+pad)
    __shared__ _Float16 Bs[128 * 40];   // 128 rows x 32 k (+pad)
    const int tid  = threadIdx.x;
    const int lane = tid & 31;
    const int wid  = tid >> 5;
    const int wm   = wid & 3;            // 4 M-subtiles (16 rows each)
    const int wn   = wid >> 2;           // 2 N-subtiles (64 wide each)
    const int m0   = blockIdx.y * 64;
    const int n0   = blockIdx.x * 128;
    const int kh   = (lane >> 4) & 1;
    const int lidx = lane & 15;

    v8f c0 = {}, c1 = {}, c2 = {}, c3 = {};

    // A tile: 2048 halfs -> 8 halfs (16B) per thread
    const int aRow = tid >> 2;           // tid*8/32
    const int aCol = (tid & 3) * 8;
    // B tile: 4096 halfs -> 16 halfs (2x16B) per thread
    const int bRow = tid >> 1;           // tid*16/32
    const int bCol = (tid & 1) * 16;

    for (int k0 = 0; k0 < K; k0 += 32) {
        __syncthreads();
        {
            const float4* sa = reinterpret_cast<const float4*>(
                A + (size_t)(m0 + aRow) * K + k0 + aCol);
            *reinterpret_cast<float4*>(&As[aRow * 40 + aCol]) = *sa;
            const float4* sb = reinterpret_cast<const float4*>(
                Bw + (size_t)(n0 + bRow) * K + k0 + bCol);
            *reinterpret_cast<float4*>(&Bs[bRow * 40 + bCol]) = sb[0];
            *reinterpret_cast<float4*>(&Bs[bRow * 40 + bCol + 8]) = sb[1];
            if (k0 + 32 < K) {
                __builtin_prefetch(A  + (size_t)(m0 + aRow) * K + k0 + 32 + aCol, 0, 1);
                __builtin_prefetch(Bw + (size_t)(n0 + bRow) * K + k0 + 32 + bCol, 0, 1);
            }
        }
        __syncthreads();
        v16h a  = gather_k(&As[(wm * 16 + lidx) * 40], kh);
        v16h b0 = gather_k(&Bs[(wn * 64 + lidx) * 40], kh);
        v16h b1 = gather_k(&Bs[(wn * 64 + 16 + lidx) * 40], kh);
        v16h b2 = gather_k(&Bs[(wn * 64 + 32 + lidx) * 40], kh);
        v16h b3 = gather_k(&Bs[(wn * 64 + 48 + lidx) * 40], kh);
        c0 = __builtin_amdgcn_wmma_f32_16x16x32_f16(false, a, false, b0,
                                                    (short)0, c0, false, false);
        c1 = __builtin_amdgcn_wmma_f32_16x16x32_f16(false, a, false, b1,
                                                    (short)0, c1, false, false);
        c2 = __builtin_amdgcn_wmma_f32_16x16x32_f16(false, a, false, b2,
                                                    (short)0, c2, false, false);
        c3 = __builtin_amdgcn_wmma_f32_16x16x32_f16(false, a, false, b3,
                                                    (short)0, c3, false, false);
    }

    const int nbase = n0 + wn * 64 + lidx;
#pragma unroll
    for (int j = 0; j < 8; ++j) {
        int m = m0 + wm * 16 + j + 8 * kh;
        if (storeHalf) {
            _Float16* C = (_Float16*)Cout;
            C[(size_t)m * N + nbase]      = (_Float16)c0[j];
            C[(size_t)m * N + nbase + 16] = (_Float16)c1[j];
            C[(size_t)m * N + nbase + 32] = (_Float16)c2[j];
            C[(size_t)m * N + nbase + 48] = (_Float16)c3[j];
        } else {
            float* C = (float*)Cout;
            C[(size_t)m * N + nbase]      = c0[j];
            C[(size_t)m * N + nbase + 16] = c1[j];
            C[(size_t)m * N + nbase + 32] = c2[j];
            C[(size_t)m * N + nbase + 48] = c3[j];
        }
    }
}

// --------------------------------------------------------------------------
// Flash attention, block-cooperative K/V staging.
// Block = 8 waves sharing one (b,h); wave w owns q-tile (qchunk*8 + w).
// Per 32-key step: K tile (32x64) and transposed V tile (Vt[64][32]) are
// staged in LDS with b128 loads, so every WMMA fragment (Q, K, P, V) is a
// contiguous-K gather that lowers to ds_load_b128 pairs.
__global__ void flash_attn(const _Float16* __restrict__ qh,
                           const _Float16* __restrict__ kmat,
                           const _Float16* __restrict__ vmat,
                           const int* __restrict__ amask,
                           _Float16* __restrict__ ctx) {
    __shared__ _Float16 Ks[32 * 72];        // 32 keys x 64 d (+pad)
    __shared__ _Float16 Vt[64 * 40];        // 64 d x 32 keys (+pad)
    __shared__ _Float16 Pbuf[8][16 * 40];   // per-wave P tile 16x32

    const int tid  = threadIdx.x;
    const int lane = tid & 31;
    const int wid  = tid >> 5;
    const int qchunk = blockIdx.x & (S_LEN / 128 - 1);   // 16 chunks of 128 rows
    const int bh     = blockIdx.x >> 4;
    const int h      = bh & (NHEADS - 1);
    const int b      = bh >> 4;
    const int khl = (lane >> 4) & 1;
    const int nl  = lane & 15;
    const int qtile = qchunk * 8 + wid;
    const int q0    = qtile * 16;

    const _Float16* qbase = qh + ((size_t)(b * S_LEN + q0)) * DMODEL + h * DK;
    v16h qa0 = gather_k(qbase + (size_t)nl * DMODEL, khl);       // d 0..31
    v16h qa1 = gather_k(qbase + (size_t)nl * DMODEL + 32, khl);  // d 32..63

    v8f acc0 = {}, acc1 = {}, acc2 = {}, acc3 = {};
    float mrow[8], lrow[8];
#pragma unroll
    for (int j = 0; j < 8; ++j) { mrow[j] = -1e30f; lrow[j] = 0.0f; }

    _Float16* P = &Pbuf[wid][0];

    // K staging: 32x64 tile, 8 halfs per thread
    const int kRow = tid >> 3;           // 0..31 (key)
    const int kCol = (tid & 7) * 8;      // 0..56 (d)
    // V staging uses the same mapping, then transposes into Vt.
    const int kvmax = qchunk * 128 + 127;   // block-uniform causal bound

    for (int j0 = 0; j0 <= kvmax; j0 += 32) {
        __syncthreads();
        {
            int key = j0 + kRow;
            int keyc = key < S_LEN ? key : S_LEN - 1;
            const float4* sk = reinterpret_cast<const float4*>(
                kmat + ((size_t)(b * S_LEN + keyc)) * DMODEL + h * DK + kCol);
            *reinterpret_cast<float4*>(&Ks[kRow * 72 + kCol]) = *sk;
            const float4* sv = reinterpret_cast<const float4*>(
                vmat + ((size_t)(b * S_LEN + keyc)) * DMODEL + h * DK + kCol);
            _Float16 tmp[8];
            *reinterpret_cast<float4*>(tmp) = *sv;
#pragma unroll
            for (int i = 0; i < 8; ++i)
                Vt[(kCol + i) * 40 + kRow] = tmp[i];   // transpose into Vt[d][key]
        }
        __syncthreads();

        // ---- scores S = Q K^T for two 16-key tiles ----
        v8f s0 = {}, s1 = {};
        {
            const _Float16* kb0 = &Ks[(size_t)nl * 72];
            const _Float16* kb1 = &Ks[(size_t)(16 + nl) * 72];
            v16h kf = gather_k(kb0, khl);
            s0 = __builtin_amdgcn_wmma_f32_16x16x32_f16(false, qa0, false, kf,
                                                        (short)0, s0, false, false);
            kf = gather_k(kb0 + 32, khl);
            s0 = __builtin_amdgcn_wmma_f32_16x16x32_f16(false, qa1, false, kf,
                                                        (short)0, s0, false, false);
            kf = gather_k(kb1, khl);
            s1 = __builtin_amdgcn_wmma_f32_16x16x32_f16(false, qa0, false, kf,
                                                        (short)0, s1, false, false);
            kf = gather_k(kb1 + 32, khl);
            s1 = __builtin_amdgcn_wmma_f32_16x16x32_f16(false, qa1, false, kf,
                                                        (short)0, s1, false, false);
        }
        // ---- scale + causal/pad mask (C layout: n = nl, m = j + 8*khl) ----
        int key0 = j0 + nl;
        int key1 = j0 + 16 + nl;
        int pm0 = (key0 < S_LEN) ? amask[b * S_LEN + key0] : 0;
        int pm1 = (key1 < S_LEN) ? amask[b * S_LEN + key1] : 0;
        float sv0[8], sv1[8];
#pragma unroll
        for (int j = 0; j < 8; ++j) {
            int mabs = q0 + j + 8 * khl;
            sv0[j] = (pm0 && key0 <= mabs) ? s0[j] * 0.125f : NEGVAL;
            sv1[j] = (pm1 && key1 <= mabs) ? s1[j] * 0.125f : NEGVAL;
        }
        // ---- online softmax: row reductions across 16 lanes of a half ----
        float mx[8];
#pragma unroll
        for (int j = 0; j < 8; ++j) mx[j] = fmaxf(sv0[j], sv1[j]);
#pragma unroll
        for (int off = 8; off >= 1; off >>= 1)
#pragma unroll
            for (int j = 0; j < 8; ++j)
                mx[j] = fmaxf(mx[j], __shfl_xor(mx[j], off, 32));
        float alpha[8], rsum[8];
#pragma unroll
        for (int j = 0; j < 8; ++j) {
            float mnew = fmaxf(mrow[j], mx[j]);
            alpha[j] = __expf(mrow[j] - mnew);
            mrow[j] = mnew;
            float p0 = __expf(sv0[j] - mnew);
            float p1 = __expf(sv1[j] - mnew);
            sv0[j] = p0; sv1[j] = p1;
            rsum[j] = p0 + p1;
        }
#pragma unroll
        for (int off = 8; off >= 1; off >>= 1)
#pragma unroll
            for (int j = 0; j < 8; ++j)
                rsum[j] += __shfl_xor(rsum[j], off, 32);
#pragma unroll
        for (int j = 0; j < 8; ++j) {
            lrow[j] = lrow[j] * alpha[j] + rsum[j];
            acc0[j] = acc0[j] * alpha[j];
            acc1[j] = acc1[j] * alpha[j];
            acc2[j] = acc2[j] * alpha[j];
            acc3[j] = acc3[j] * alpha[j];
        }
        // ---- P: C layout -> row-major 16x32 in per-wave LDS ----
#pragma unroll
        for (int j = 0; j < 8; ++j) {
            int m = j + 8 * khl;
            P[m * 40 + nl]      = (_Float16)sv0[j];
            P[m * 40 + 16 + nl] = (_Float16)sv1[j];
        }
        // wave-internal LDS RAW: wait for our own ds stores
        asm volatile("s_wait_dscnt 0x0" ::: "memory");
        v16h pa = gather_k(P + nl * 40, khl);
        // ---- acc += P @ V: V frags from transposed Vt, contiguous-K ----
        v16h vf;
        vf = gather_k(&Vt[(size_t)(0  + nl) * 40], khl);
        acc0 = __builtin_amdgcn_wmma_f32_16x16x32_f16(false, pa, false, vf,
                                                      (short)0, acc0, false, false);
        vf = gather_k(&Vt[(size_t)(16 + nl) * 40], khl);
        acc1 = __builtin_amdgcn_wmma_f32_16x16x32_f16(false, pa, false, vf,
                                                      (short)0, acc1, false, false);
        vf = gather_k(&Vt[(size_t)(32 + nl) * 40], khl);
        acc2 = __builtin_amdgcn_wmma_f32_16x16x32_f16(false, pa, false, vf,
                                                      (short)0, acc2, false, false);
        vf = gather_k(&Vt[(size_t)(48 + nl) * 40], khl);
        acc3 = __builtin_amdgcn_wmma_f32_16x16x32_f16(false, pa, false, vf,
                                                      (short)0, acc3, false, false);
    }

    // ---- normalize + store ctx (f16, [B,S,H*DK]) ----
#pragma unroll
    for (int j = 0; j < 8; ++j) {
        int m = q0 + j + 8 * khl;
        float inv = 1.0f / fmaxf(lrow[j], 1e-20f);
        _Float16* o = ctx + ((size_t)(b * S_LEN + m)) * DMODEL + h * DK + nl;
        o[0]  = (_Float16)(acc0[j] * inv);
        o[16] = (_Float16)(acc1[j] * inv);
        o[32] = (_Float16)(acc2[j] * inv);
        o[48] = (_Float16)(acc3[j] * inv);
    }
}

// --------------------------------------------------------------------------
extern "C" void kernel_launch(void* const* d_in, const int* in_sizes, int n_in,
                              void* d_out, int out_size, void* d_ws, size_t ws_size,
                              hipStream_t stream) {
    const float* x     = (const float*)d_in[0];
    const int*   amask = (const int*)d_in[1];
    const float* wq    = (const float*)d_in[2];
    const float* wk    = (const float*)d_in[3];
    const float* wv    = (const float*)d_in[4];
    const float* wo    = (const float*)d_in[5];

    const int M = BATCH * S_LEN;   // 4096
    const int K = DMODEL;          // 1024
    const int N = DMODEL;          // 1024

    char* ws = (char*)d_ws;
    size_t off = 0;
    auto alloc = [&](size_t bytes) -> _Float16* {
        _Float16* p = (_Float16*)(ws + off);
        off += (bytes + 255) & ~(size_t)255;
        return p;
    };
    _Float16* xh   = alloc((size_t)M * K * 2);
    _Float16* wqh  = alloc((size_t)N * K * 2);
    _Float16* wkh  = alloc((size_t)N * K * 2);
    _Float16* wvh  = alloc((size_t)N * K * 2);
    _Float16* woh  = alloc((size_t)N * K * 2);
    _Float16* qh   = alloc((size_t)M * N * 2);
    _Float16* khb  = alloc((size_t)M * N * 2);
    _Float16* vhb  = alloc((size_t)M * N * 2);
    _Float16* ctxh = alloc((size_t)M * N * 2);

    const int nx = M * K;
    const int nw = N * K;
    cvt_f32_to_f16<<<(nx + 255) / 256, 256, 0, stream>>>(x,  xh,  nx);
    cvt_f32_to_f16<<<(nw + 255) / 256, 256, 0, stream>>>(wq, wqh, nw);
    cvt_f32_to_f16<<<(nw + 255) / 256, 256, 0, stream>>>(wk, wkh, nw);
    cvt_f32_to_f16<<<(nw + 255) / 256, 256, 0, stream>>>(wv, wvh, nw);
    cvt_f32_to_f16<<<(nw + 255) / 256, 256, 0, stream>>>(wo, woh, nw);

    dim3 gblk(256);
    dim3 ggrid(N / 128, M / 64);
    wmma_gemm_nt<<<ggrid, gblk, 0, stream>>>(xh, wqh, qh,  M, N, K, 1);
    wmma_gemm_nt<<<ggrid, gblk, 0, stream>>>(xh, wkh, khb, M, N, K, 1);
    wmma_gemm_nt<<<ggrid, gblk, 0, stream>>>(xh, wvh, vhb, M, N, K, 1);

    const int nr = BATCH * S_LEN * NHEADS * 32;
    rope_inplace<<<(nr + 255) / 256, 256, 0, stream>>>(qh, khb, nr);

    const int nblocks = BATCH * NHEADS * (S_LEN / 128);   // 512 blocks
    flash_attn<<<nblocks, 256, 0, stream>>>(qh, khb, vhb, amask, ctxh);

    wmma_gemm_nt<<<ggrid, gblk, 0, stream>>>(ctxh, woh, d_out, M, N, K, 0);
}
